// MoEBlockLayer_77257871720878
// MI455X (gfx1250) — matrix operand order repaired and checked by reading, not compiled, not run on hardware
//
#include <hip/hip_runtime.h>

// ---------------- problem constants (match reference) ----------------
constexpr int kB = 4, kT = 2048, kD = 2048, kE = 4, kH = 8192;
constexpr int kN   = kB * kT;   // 8192 tokens
constexpr int kCAP = kN / kE;   // 2048 capacity per expert (CAPACITY_FACTOR=1)

// ---------------- types ----------------
typedef __attribute__((ext_vector_type(16))) __bf16        v16bf;
typedef __attribute__((ext_vector_type(8)))  float         v8f;
typedef __attribute__((ext_vector_type(4)))  unsigned int  u32x4;
typedef __attribute__((ext_vector_type(4)))  float         f32x4;

union FragU { v16bf v; u32x4 u[2]; };

__device__ inline unsigned short f32_to_bf16(float f) {
  unsigned int u = __float_as_uint(f);
  return (unsigned short)((u + 0x7FFFu + ((u >> 16) & 1u)) >> 16);  // RNE
}
__device__ inline unsigned int pack_bf16x2(float lo, float hi) {
  return (unsigned int)f32_to_bf16(lo) | ((unsigned int)f32_to_bf16(hi) << 16);
}
__device__ inline float gelu_exact(float v) {
  return 0.5f * v * (1.0f + erff(v * 0.70710678118654752f));
}
// low 32 bits of a generic pointer to __shared__ data == LDS byte offset
__device__ inline unsigned lds_off(const void* p) { return (unsigned)(uintptr_t)p; }

// ---------------- kernel 0: zero output / routing scratch ----------------
__global__ void moe_init(float* __restrict__ out, int* __restrict__ cnt,
                         int* __restrict__ tok_list, float* __restrict__ p_list) {
  const size_t i = (size_t)blockIdx.x * blockDim.x + threadIdx.x;
  if (i < (size_t)kN * kD) out[i] = 0.0f;
  if (i < (size_t)kE) cnt[i] = 0;
  if (i < (size_t)kE * kCAP) { tok_list[i] = -1; p_list[i] = 0.0f; }
}

// ---------------- kernel 1: router (one wave32 per token) ----------------
__global__ __launch_bounds__(256) void moe_router(
    const float* __restrict__ x, const float* __restrict__ Wg,
    const float* __restrict__ bg, int* __restrict__ top_idx,
    float* __restrict__ top_p) {
  const int t    = (blockIdx.x * blockDim.x + threadIdx.x) >> 5;
  const int lane = threadIdx.x & 31;
  if (t >= kN) return;
  const float* __restrict__ xr = x + (size_t)t * kD;
  float a0 = 0.f, a1 = 0.f, a2 = 0.f, a3 = 0.f;
  for (int d = lane; d < kD; d += 32) {
    const float xv = xr[d];
    const f32x4 w = *(const f32x4*)(Wg + (size_t)d * kE);  // Wg row = 4 contig floats
    a0 += xv * w.x; a1 += xv * w.y; a2 += xv * w.z; a3 += xv * w.w;
  }
#pragma unroll
  for (int off = 16; off > 0; off >>= 1) {
    a0 += __shfl_xor(a0, off, 32);
    a1 += __shfl_xor(a1, off, 32);
    a2 += __shfl_xor(a2, off, 32);
    a3 += __shfl_xor(a3, off, 32);
  }
  if (lane == 0) {
    float l[4] = {a0 + bg[0], a1 + bg[1], a2 + bg[2], a3 + bg[3]};
    int i0 = 0;
    for (int j = 1; j < 4; ++j) if (l[j] > l[i0]) i0 = j;        // ties -> lowest idx
    int i1 = (i0 == 0) ? 1 : 0;
    for (int j = 0; j < 4; ++j) if (j != i0 && l[j] > l[i1]) i1 = j;
    const float e1  = __expf(l[i1] - l[i0]);                      // stable 2-way softmax
    const float inv = 1.0f / (1.0f + e1);
    top_idx[2 * t] = i0; top_idx[2 * t + 1] = i1;
    top_p[2 * t]   = inv; top_p[2 * t + 1]  = e1 * inv;
  }
}

// ---------------- kernel 2: capacity-limited slot assignment ----------------
__global__ void moe_assign(const int* __restrict__ top_idx,
                           const float* __restrict__ top_p, int* __restrict__ cnt,
                           int* __restrict__ tok_list, float* __restrict__ p_list) {
  const int t = blockIdx.x * blockDim.x + threadIdx.x;
  if (t >= kN) return;
#pragma unroll
  for (int k = 0; k < 2; ++k) {
    const int e   = top_idx[2 * t + k];
    const float p = top_p[2 * t + k];
    const int slot = atomicAdd(&cnt[e], 1);
    if (slot < kCAP) {
      tok_list[e * kCAP + slot] = t;
      p_list[e * kCAP + slot]   = p;
    }
  }
}

// ---------------- grouped-GEMM tiling ----------------
#define BM 128
#define BN 128
#define BK 64
#define LDSK (BK + 8)   // row stride 144B (multiple of 16B)

// kernel 3: h[e] = gelu(x[tok_list[e]] @ W1[e])   (bf16 WMMA, f32 accumulate)
__global__ __launch_bounds__(256) void moe_gemm1_gelu(
    const float* __restrict__ x, const float* __restrict__ W1,
    const int* __restrict__ tok_list, unsigned short* __restrict__ hbuf) {
  __shared__ __align__(16) unsigned short ldsA[BM][LDSK];
  __shared__ __align__(16) unsigned short ldsB[BN][LDSK];  // N-major (pre-transposed)

  const int e = blockIdx.z;
  const int mBlk = blockIdx.y * BM, nBlk = blockIdx.x * BN;
  const int tid = threadIdx.x, lane = tid & 31, wave = tid >> 5;
  const int waveM = (wave >> 2) * 64, waveN = (wave & 3) * 32;
  const int half = lane >> 4, l15 = lane & 15;

  // A staging: thread -> (row ar, 32 K-elems at ak)
  const int ar = tid >> 1, ak = (tid & 1) * 32;
  const int tokA = tok_list[e * kCAP + mBlk + ar];
  const float* __restrict__ arow = x + (size_t)(tokA < 0 ? 0 : tokA) * kD;
  // B staging: thread -> (col bn, 32 K-elems at bk)
  const int bn = tid & 127, bk = (tid >> 7) * 32;
  const float* __restrict__ We = W1 + (size_t)e * kD * kH;  // [D][H] row-major

  v8f acc[4][2] = {};

  for (int k0 = 0; k0 < kD; k0 += BK) {
    // ---- A tile: gather token row, f32 -> bf16, vector LDS stores ----
    f32x4 f[8] = {};
    if (tokA >= 0) {
      const f32x4* s = (const f32x4*)(arow + k0 + ak);
#pragma unroll
      for (int j = 0; j < 8; ++j) f[j] = s[j];
    }
#pragma unroll
    for (int j = 0; j < 4; ++j) {
      u32x4 pA;
      pA.x = pack_bf16x2(f[2 * j].x, f[2 * j].y);
      pA.y = pack_bf16x2(f[2 * j].z, f[2 * j].w);
      pA.z = pack_bf16x2(f[2 * j + 1].x, f[2 * j + 1].y);
      pA.w = pack_bf16x2(f[2 * j + 1].z, f[2 * j + 1].w);
      *(u32x4*)&ldsA[ar][ak + 8 * j] = pA;
    }
    // ---- B tile: coalesced f32 loads, pack pairs, vector LDS stores (N-major) ----
    unsigned int pk[16];
#pragma unroll
    for (int i = 0; i < 16; ++i) {
      const float lo = We[(size_t)(k0 + bk + 2 * i)     * kH + nBlk + bn];
      const float hi = We[(size_t)(k0 + bk + 2 * i + 1) * kH + nBlk + bn];
      pk[i] = pack_bf16x2(lo, hi);
    }
#pragma unroll
    for (int j = 0; j < 4; ++j) {
      u32x4 pB; pB.x = pk[4*j]; pB.y = pk[4*j+1]; pB.z = pk[4*j+2]; pB.w = pk[4*j+3];
      *(u32x4*)&ldsB[bn][bk + 8 * j] = pB;
    }
    if (k0 + BK < kD)  // stream next weight tile toward L2/WGP$
      __builtin_prefetch(&We[(size_t)(k0 + BK + bk) * kH + nBlk + bn], 0, 1);
    __syncthreads();

    // ---- fragments + 16 WMMAs per wave per stage ----
#pragma unroll
    for (int ks = 0; ks < BK; ks += 32) {
      FragU fa[4], fb[2];
#pragma unroll
      for (int mi = 0; mi < 4; ++mi) {
        const unsigned short* base = &ldsA[waveM + mi * 16 + l15][ks + half * 8];
        fa[mi].u[0] = *(const u32x4*)base;          // K 0..7   / 8..15
        fa[mi].u[1] = *(const u32x4*)(base + 16);   // K 16..23 / 24..31
      }
#pragma unroll
      for (int ni = 0; ni < 2; ++ni) {
        const unsigned short* base = &ldsB[waveN + ni * 16 + l15][ks + half * 8];
        fb[ni].u[0] = *(const u32x4*)base;          // K 0..7   / 8..15
        fb[ni].u[1] = *(const u32x4*)(base + 16);   // K 16..23 / 24..31
      }
#pragma unroll
      for (int mi = 0; mi < 4; ++mi)
#pragma unroll
        for (int ni = 0; ni < 2; ++ni)
          acc[mi][ni] = __builtin_amdgcn_wmma_f32_16x16x32_bf16(
              false, fa[mi].v, false, fb[ni].v, (short)0, acc[mi][ni], false, false);
    }
    __syncthreads();
  }

  // ---- epilogue: exact GELU, store h as bf16 ----
#pragma unroll
  for (int mi = 0; mi < 4; ++mi) {
    const int mBase = mBlk + waveM + mi * 16 + half * 8;
#pragma unroll
    for (int ni = 0; ni < 2; ++ni) {
      const int n = nBlk + waveN + ni * 16 + l15;
#pragma unroll
      for (int i = 0; i < 8; ++i) {
        const int m = mBase + i;
        hbuf[((size_t)e * kCAP + m) * kH + n] = f32_to_bf16(gelu_exact(acc[mi][ni][i]));
      }
    }
  }
}

// kernel 4: out[tok] += p * (h[e] @ W2[e])
// A operand is raw bf16 -> stream it with CDNA5 async global->LDS copies.
__global__ __launch_bounds__(256) void moe_gemm2_scatter(
    const unsigned short* __restrict__ hbuf, const float* __restrict__ W2,
    const int* __restrict__ tok_list, const float* __restrict__ p_list,
    float* __restrict__ out) {
  __shared__ __align__(16) unsigned short ldsA[BM][LDSK];
  __shared__ __align__(16) unsigned short ldsB[BN][LDSK];

  const int e = blockIdx.z;
  const int mBlk = blockIdx.y * BM, nBlk = blockIdx.x * BN;
  const int tid = threadIdx.x, lane = tid & 31, wave = tid >> 5;
  const int waveM = (wave >> 2) * 64, waveN = (wave & 3) * 32;
  const int half = lane >> 4, l15 = lane & 15;

  const int ar = tid >> 1, ak = (tid & 1) * 32;
  const unsigned short* __restrict__ hrow =
      hbuf + ((size_t)e * kCAP + mBlk + ar) * kH;
  const unsigned aLds = lds_off(&ldsA[ar][ak]);     // LDS byte offset for async dst
  const int bn = tid & 127, bk = (tid >> 7) * 32;
  const float* __restrict__ We = W2 + (size_t)e * kH * kD;  // [H][D] row-major

  v8f acc[4][2] = {};

  for (int k0 = 0; k0 < kH; k0 += BK) {
    // ---- A tile: async copy 64B/thread global->LDS (no VGPR round-trip) ----
    {
      const unsigned short* g = hrow + k0 + ak;
      // INST_OFFSET is added to BOTH the LDS and the global address (ISA 08 §4.4)
      asm volatile(
          "global_load_async_to_lds_b128 %0, %1, off\n\t"
          "global_load_async_to_lds_b128 %0, %1, off offset:16\n\t"
          "global_load_async_to_lds_b128 %0, %1, off offset:32\n\t"
          "global_load_async_to_lds_b128 %0, %1, off offset:48"
          :: "v"(aLds), "v"(g) : "memory");
    }
    // ---- B tile: coalesced f32 loads, pack pairs, vector LDS stores ----
    unsigned int pk[16];
#pragma unroll
    for (int i = 0; i < 16; ++i) {
      const float lo = We[(size_t)(k0 + bk + 2 * i)     * kD + nBlk + bn];
      const float hi = We[(size_t)(k0 + bk + 2 * i + 1) * kD + nBlk + bn];
      pk[i] = pack_bf16x2(lo, hi);
    }
#pragma unroll
    for (int j = 0; j < 4; ++j) {
      u32x4 pB; pB.x = pk[4*j]; pB.y = pk[4*j+1]; pB.z = pk[4*j+2]; pB.w = pk[4*j+3];
      *(u32x4*)&ldsB[bn][bk + 8 * j] = pB;
    }
    if (k0 + BK < kH)
      __builtin_prefetch(&We[(size_t)(k0 + BK + bk) * kD + nBlk + bn], 0, 1);
    asm volatile("s_wait_asynccnt 0x0" ::: "memory");  // async LDS writes visible
    __syncthreads();

#pragma unroll
    for (int ks = 0; ks < BK; ks += 32) {
      FragU fa[4], fb[2];
#pragma unroll
      for (int mi = 0; mi < 4; ++mi) {
        const unsigned short* base = &ldsA[waveM + mi * 16 + l15][ks + half * 8];
        fa[mi].u[0] = *(const u32x4*)base;
        fa[mi].u[1] = *(const u32x4*)(base + 16);
      }
#pragma unroll
      for (int ni = 0; ni < 2; ++ni) {
        const unsigned short* base = &ldsB[waveN + ni * 16 + l15][ks + half * 8];
        fb[ni].u[0] = *(const u32x4*)base;
        fb[ni].u[1] = *(const u32x4*)(base + 16);
      }
#pragma unroll
      for (int mi = 0; mi < 4; ++mi)
#pragma unroll
        for (int ni = 0; ni < 2; ++ni)
          acc[mi][ni] = __builtin_amdgcn_wmma_f32_16x16x32_bf16(
              false, fa[mi].v, false, fb[ni].v, (short)0, acc[mi][ni], false, false);
    }
    __syncthreads();
  }

  // ---- epilogue: scale by gate prob, atomic scatter-add (token may be in 2 experts) ----
#pragma unroll
  for (int mi = 0; mi < 4; ++mi) {
#pragma unroll
    for (int i = 0; i < 8; ++i) {
      const int m = mBlk + waveM + mi * 16 + half * 8 + i;
      const int tok = tok_list[e * kCAP + m];
      const float pp = p_list[e * kCAP + m];
      if (tok >= 0) {
        float* __restrict__ orow = out + (size_t)tok * kD;
#pragma unroll
        for (int ni = 0; ni < 2; ++ni)
          atomicAdd(orow + (nBlk + waveN + ni * 16 + l15), acc[mi][ni][i] * pp);
      }
    }
  }
}

// ---------------- workspace layout (bytes) ----------------
constexpr size_t OFF_CNT  = 0;                                 // 4 ints
constexpr size_t OFF_TOK  = 256;                               // E*CAP ints  (32 KB)
constexpr size_t OFF_P    = OFF_TOK  + (size_t)kE * kCAP * 4;  // E*CAP f32   (32 KB)
constexpr size_t OFF_TIDX = OFF_P    + (size_t)kE * kCAP * 4;  // N*2 ints    (64 KB)
constexpr size_t OFF_TP   = OFF_TIDX + (size_t)kN * 2 * 4;     // N*2 f32     (64 KB)
constexpr size_t OFF_H    = OFF_TP   + (size_t)kN * 2 * 4;     // E*CAP*H bf16 (128 MB)

extern "C" void kernel_launch(void* const* d_in, const int* in_sizes, int n_in,
                              void* d_out, int out_size, void* d_ws, size_t ws_size,
                              hipStream_t stream) {
  const float* x  = (const float*)d_in[0];
  const float* Wg = (const float*)d_in[1];
  const float* bg = (const float*)d_in[2];
  const float* W1 = (const float*)d_in[3];
  const float* W2 = (const float*)d_in[4];
  float* out = (float*)d_out;

  char* ws = (char*)d_ws;
  int*   cnt      = (int*)(ws + OFF_CNT);
  int*   tok_list = (int*)(ws + OFF_TOK);
  float* p_list   = (float*)(ws + OFF_P);
  int*   top_idx  = (int*)(ws + OFF_TIDX);
  float* top_p    = (float*)(ws + OFF_TP);
  unsigned short* hbuf = (unsigned short*)(ws + OFF_H);

  moe_init<<<((size_t)kN * kD + 255) / 256, 256, 0, stream>>>(out, cnt, tok_list, p_list);
  moe_router<<<(kN * 32 + 255) / 256, 256, 0, stream>>>(x, Wg, bg, top_idx, top_p);
  moe_assign<<<(kN + 255) / 256, 256, 0, stream>>>(top_idx, top_p, cnt, tok_list, p_list);
  moe_gemm1_gelu<<<dim3(kH / BN, kCAP / BM, kE), 256, 0, stream>>>(x, W1, tok_list, hbuf);
  moe_gemm2_scatter<<<dim3(kD / BN, kCAP / BM, kE), 256, 0, stream>>>(hbuf, W2, tok_list, p_list, out);
}
